// DGCNN_11854109737377
// MI455X (gfx1250) — matrix operand order repaired and checked by reading, compiled, not run on hardware
//
#include <hip/hip_runtime.h>
#include <hip/hip_bf16.h>

typedef __attribute__((ext_vector_type(16))) __bf16 v16bf;
typedef __attribute__((ext_vector_type(8)))  __bf16 v8bf;
typedef __attribute__((ext_vector_type(8)))  float  v8f;
typedef __attribute__((ext_vector_type(2)))  float  v2f;
typedef unsigned int uint32_tt;
typedef __attribute__((ext_vector_type(4))) uint32_tt u32x4;
typedef __attribute__((ext_vector_type(8))) uint32_tt u32x8;

#define NB 8
#define NP 1024
#define KNN 20
#define NNODES (NB * NP)        // 8192
#define NEDGES (NNODES * KNN)   // 163840
#define BN_EPS 1e-5f

// ---------------------------------------------------------------- sq norms
__global__ void sqnorm_kernel(const float* __restrict__ X, float* __restrict__ sq, int Cin) {
    int t = blockIdx.x * blockDim.x + threadIdx.x;
    if (t >= NNODES) return;
    float s = 0.f;
    for (int c = 0; c < Cin; ++c) { float v = X[(size_t)t * Cin + c]; s += v * v; }
    sq[t] = s;
}

// ------------------------------------------------- Gram matrix via f32 WMMA
template <int CIN>
__global__ void gram_wmma(const float* __restrict__ X, float* __restrict__ G) {
    int lane = threadIdx.x;
    int tn = blockIdx.x, tm = blockIdx.y, b = blockIdx.z;
    int half = lane >> 4, l16 = lane & 15;
    int arow = tm * 16 + l16;
    int bcol = tn * 16 + l16;
    const float* Xb = X + (size_t)b * NP * CIN;
    v8f acc = {};
    constexpr int KP4 = (CIN + 3) & ~3;
#pragma unroll
    for (int k0 = 0; k0 < KP4; k0 += 4) {
        int kb = k0 + half * 2;
        v2f av, bv;
        if constexpr ((CIN & 3) == 0) {
            av = *(const v2f*)(Xb + (size_t)arow * CIN + kb);
            bv = *(const v2f*)(Xb + (size_t)bcol * CIN + kb);
        } else {
            int i0 = (kb     < CIN) ? kb     : 0;
            int i1 = (kb + 1 < CIN) ? kb + 1 : 0;
            float a0 = Xb[(size_t)arow * CIN + i0];
            float a1 = Xb[(size_t)arow * CIN + i1];
            float b0 = Xb[(size_t)bcol * CIN + i0];
            float b1 = Xb[(size_t)bcol * CIN + i1];
            av.x = (kb     < CIN) ? a0 : 0.f;
            av.y = (kb + 1 < CIN) ? a1 : 0.f;
            bv.x = (kb     < CIN) ? b0 : 0.f;
            bv.y = (kb + 1 < CIN) ? b1 : 0.f;
        }
        acc = __builtin_amdgcn_wmma_f32_16x16x4_f32(false, av, false, bv,
                                                    (short)0, acc, false, false);
    }
    float* Gb = G + (size_t)b * NP * NP;
#pragma unroll
    for (int v = 0; v < 8; ++v) {
        int row = tm * 16 + v + half * 8;
        Gb[(size_t)row * NP + tn * 16 + l16] = acc[v];
    }
}

// ------------------------------------------------------------- KNN top-k
__global__ void knn_topk(const float* __restrict__ G, const float* __restrict__ sq,
                         int* __restrict__ knn) {
    int t = blockIdx.x * blockDim.x + threadIdx.x;
    if (t >= NNODES) return;
    int b = t >> 10, p = t & 1023;
    const float* Gr  = G + ((size_t)b * NP + p) * NP;
    const float* sqb = sq + b * NP;
    float sp = sqb[p];
    float best[KNN]; int bidx[KNN];
#pragma unroll
    for (int i = 0; i < KNN; ++i) { best[i] = 3.4e38f; bidx[i] = 0; }
    for (int q = 0; q < NP; ++q) {
        if (q == p) continue;
        float d = sp + sqb[q] - 2.f * Gr[q];
        if (d < best[KNN - 1]) {
            int pos = KNN - 1;
            while (pos > 0 && best[pos - 1] > d) {
                best[pos] = best[pos - 1]; bidx[pos] = bidx[pos - 1]; --pos;
            }
            best[pos] = d; bidx[pos] = q;
        }
    }
    for (int i = 0; i < KNN; ++i) knn[(size_t)t * KNN + i] = bidx[i];
}

// ----------------------------------------------- edge features -> bf16, padded
__global__ void gather_feat(const float* __restrict__ X, const int* __restrict__ knn,
                            __bf16* __restrict__ F, int Cin) {
    int e = blockIdx.x;
    int c = threadIdx.x;
    int Kp = blockDim.x;
    int node = e / KNN;
    int b = node >> 10;
    int j = knn[e];
    int nj = (b << 10) + j;
    float v = 0.f;
    if (c < Cin)          v = X[(size_t)node * Cin + c] - X[(size_t)nj * Cin + c];
    else if (c < 2 * Cin) v = X[(size_t)nj * Cin + (c - Cin)];
    F[(size_t)e * Kp + c] = (__bf16)v;
}

// ------------------------------ W -> bf16, TRANSPOSED [Cout][Kp], zero-padded k
__global__ void wconv_t(const float* __restrict__ W, __bf16* __restrict__ Wt,
                        int K2, int Kp, int Cout) {
    int t = blockIdx.x * blockDim.x + threadIdx.x;
    if (t >= Kp * Cout) return;
    int n = t / Kp, k = t % Kp;
    Wt[t] = (k < K2) ? (__bf16)W[(size_t)k * Cout + n] : (__bf16)0.f;
}

__global__ void zero_f32(float* p, int n) {
    int t = blockIdx.x * blockDim.x + threadIdx.x;
    if (t < n) p[t] = 0.f;
}

// ---------------------------------------------------- WMMA operand loaders
__device__ inline v16bf load_a(const __bf16* __restrict__ F, int Kp, int row,
                               int k0, int half) {
    const __bf16* base = F + (size_t)row * Kp + k0 + half * 8;
    v8bf lo = *(const v8bf*)(base);
    v8bf hi = *(const v8bf*)(base + 16);
    v16bf a;
#pragma unroll
    for (int s = 0; s < 8; ++s) { a[s] = lo[s]; a[s + 8] = hi[s]; }
    return a;
}
// B operand from transposed, row-padded LDS panel: two 8-element chunks per lane (k = 8h+s, 16+8h+s)
__device__ inline v16bf load_b_lds(const __bf16* __restrict__ Wl, int lstride,
                                   int k0, int lcol, int half) {
    const __bf16* base = Wl + lcol * lstride + k0 + half * 8;
    v8bf lo = *(const v8bf*)(base);
    v8bf hi = *(const v8bf*)(base + 16);
    v16bf b;
#pragma unroll
    for (int s = 0; s < 8; ++s) { b[s] = lo[s]; b[s + 8] = hi[s]; }
    return b;
}

// ------------------------------------------------ edge GEMM (two passes)
// Block: 5 waves x 32 lanes; wave w owns rows [blk*80 + w*16, +16), 64 cols.
// Transposed W panel [64 x KP] is DMA'd to LDS by the TDM with row padding
// (pad_amount = 4 DWORDs per row) for conflict-free ds_load_b128.
template <int STATS, int KP, int COUT>
__global__ __launch_bounds__(160)
void edge_gemm(const __bf16* __restrict__ F, const __bf16* __restrict__ Wt,
               const float* __restrict__ bias, const float* __restrict__ gamma,
               const float* __restrict__ beta, float* __restrict__ ssum,
               float* __restrict__ ssq, float* __restrict__ xout) {
    constexpr int LSTRIDE = KP + 8;                       // halves per padded LDS row
    constexpr uint32_tt PADI = (KP == 32) ? 3u : (KP == 64) ? 4u : (KP == 128) ? 5u : 6u;
    __shared__ __bf16 Wl[64 * LSTRIDE];
    __shared__ float tile[80 * 64];
    __shared__ float lsum[64], lsq[64];
    int lane = threadIdx.x, wave = threadIdx.y;
    int half = lane >> 4, l16 = lane & 15;
    int tid = wave * 32 + lane;
    int rows0 = blockIdx.x * 80 + wave * 16;
    int nbase = blockIdx.y * 64;

    if (STATS) {
        if (tid < 64) { lsum[tid] = 0.f; lsq[tid] = 0.f; }
    }

    // ---- TDM: DMA Wt[nbase:nbase+64, 0:KP] (bf16, rows contiguous) to LDS, padded
    if (wave == 0) {
        unsigned long long ga = (unsigned long long)(uintptr_t)(Wt + (size_t)nbase * KP);
        uint32_tt lds_off = (uint32_tt)(uintptr_t)(&Wl[0]);  // low 32 bits = LDS byte offset
        u32x4 g0;
        g0[0] = 1u;                                          // count=1
        g0[1] = lds_off;                                     // lds_addr
        g0[2] = (uint32_tt)ga;                               // global_addr[31:0]
        g0[3] = (uint32_tt)((ga >> 32) & 0x01FFFFFFu) | (2u << 30);  // addr[56:32] | type=2
        u32x8 g1;
        g1[0] = (1u << 16)                                   // data_size=1 (2B)
              | (1u << 20)                                   // pad_enable
              | (PADI << 22)                                 // pad_interval = KP/2 DWORDs
              | (3u << 25);                                  // pad_amount = 4 DWORDs
        g1[1] = ((uint32_tt)KP & 0xFFFFu) << 16;             // tensor_dim0[15:0] = KP
        g1[2] = (((uint32_tt)COUT & 0xFFFFu) << 16);         // dim0[31:16]=0 | tensor_dim1=COUT
        g1[3] = ((uint32_tt)KP << 16);                       // dim1[31:16]=0 | tile_dim0=KP
        g1[4] = 64u;                                         // tile_dim1=64, tile_dim2=0
        g1[5] = (uint32_tt)KP;                               // tensor_dim0_stride = KP
        g1[6] = 0u;
        g1[7] = 0u;
        asm volatile("tensor_load_to_lds %0, %1" :: "s"(g0), "s"(g1) : "memory");
        __builtin_amdgcn_s_wait_tensorcnt(0);
    }
    __syncthreads();

    v8f acc0 = {}, acc1 = {}, acc2 = {}, acc3 = {};
    int arow = rows0 + l16;
#pragma unroll
    for (int k0 = 0; k0 < KP; k0 += 32) {
        v16bf a  = load_a(F, KP, arow, k0, half);
        v16bf b0 = load_b_lds(Wl, LSTRIDE, k0, 0  + l16, half);
        acc0 = __builtin_amdgcn_wmma_f32_16x16x32_bf16(false, a, false, b0, (short)0, acc0, false, false);
        v16bf b1 = load_b_lds(Wl, LSTRIDE, k0, 16 + l16, half);
        acc1 = __builtin_amdgcn_wmma_f32_16x16x32_bf16(false, a, false, b1, (short)0, acc1, false, false);
        v16bf b2 = load_b_lds(Wl, LSTRIDE, k0, 32 + l16, half);
        acc2 = __builtin_amdgcn_wmma_f32_16x16x32_bf16(false, a, false, b2, (short)0, acc2, false, false);
        v16bf b3 = load_b_lds(Wl, LSTRIDE, k0, 48 + l16, half);
        acc3 = __builtin_amdgcn_wmma_f32_16x16x32_bf16(false, a, false, b3, (short)0, acc3, false, false);
    }
    const float invN = 1.f / (float)NEDGES;
#define PROC_TILE(ACC, T)                                                          \
    {                                                                              \
        int lcol = (T)*16 + l16;                                                   \
        int col  = nbase + lcol;                                                   \
        float bsv = bias[col];                                                     \
        if (STATS) {                                                               \
            float hs = 0.f, hq = 0.f;                                              \
            _Pragma("unroll") for (int v = 0; v < 8; ++v) {                        \
                float h = (ACC)[v] + bsv; hs += h; hq += h * h;                    \
            }                                                                      \
            atomicAdd(&lsum[lcol], hs);                                            \
            atomicAdd(&lsq[lcol], hq);                                             \
        } else {                                                                   \
            float mu  = ssum[col] * invN;                                          \
            float var = ssq[col] * invN - mu * mu;                                 \
            float inv = rsqrtf(var + BN_EPS);                                      \
            float g = gamma[col], be = beta[col];                                  \
            _Pragma("unroll") for (int v = 0; v < 8; ++v) {                        \
                float h  = (ACC)[v] + bsv;                                         \
                float hn = g * (h - mu) * inv + be;                                \
                hn = fmaxf(hn, 0.f);                                               \
                int rl = wave * 16 + v + half * 8;                                 \
                tile[rl * 64 + lcol] = hn;                                         \
            }                                                                      \
        }                                                                          \
    }
    PROC_TILE(acc0, 0)
    PROC_TILE(acc1, 1)
    PROC_TILE(acc2, 2)
    PROC_TILE(acc3, 3)
#undef PROC_TILE
    __syncthreads();
    if (STATS) {
        if (tid < 64) {
            atomicAdd(&ssum[nbase + tid], lsum[tid]);
            atomicAdd(&ssq[nbase + tid], lsq[tid]);
        }
    } else {
        for (int o = tid; o < 256; o += 160) {
            int nl = o >> 6, c = o & 63;
            float s = 0.f;
#pragma unroll
            for (int j = 0; j < KNN; ++j) s += tile[(nl * KNN + j) * 64 + c];
            int node = blockIdx.x * 4 + nl;
            xout[(size_t)node * COUT + nbase + c] = s;
        }
    }
}

// ---------------------------------------------------------------- tail ops
__global__ void maxpool_kernel(const float* __restrict__ X, float* __restrict__ pooled) {
    int t = blockIdx.x * blockDim.x + threadIdx.x;
    if (t >= NB * 1024) return;
    int b = t >> 10, c = t & 1023;
    float m = -3.4e38f;
    for (int p = 0; p < NP; ++p)
        m = fmaxf(m, X[((size_t)(b << 10) + p) * 1024 + c]);
    pooled[t] = m;
}
__global__ void fc1_kernel(const float* __restrict__ pooled, const float* __restrict__ W,
                           const float* __restrict__ bias, float* __restrict__ H) {
    int t = blockIdx.x * blockDim.x + threadIdx.x;
    if (t >= NB * 512) return;
    int b = t >> 9, j = t & 511;
    float s = bias[j];
    for (int c = 0; c < 1024; ++c) s += pooled[b * 1024 + c] * W[c * 512 + j];
    H[t] = fmaxf(s, 0.f);
}
__global__ void fc2_kernel(const float* __restrict__ H, const float* __restrict__ W,
                           const float* __restrict__ bias, float* __restrict__ out) {
    int t = blockIdx.x * blockDim.x + threadIdx.x;
    if (t >= NB * 40) return;
    int b = t / 40, j = t % 40;
    float s = bias[j];
    for (int c = 0; c < 512; ++c) s += H[b * 512 + c] * W[c * 40 + j];
    out[t] = s;
}

// ---------------------------------------------------------------- per-layer driver
struct Scratch {
    float* gram; float* sq; int* knn;
    __bf16* feat; __bf16* wbf;
    float* ssum; float* ssq;
};

template <int CIN, int COUT>
static void run_layer(const float* xcur, float* xnext,
                      const float* W, const float* bias, const float* g, const float* be,
                      const Scratch& S, hipStream_t stream) {
    constexpr int KP = ((2 * CIN + 31) / 32) * 32;
    sqnorm_kernel<<<(NNODES + 255) / 256, 256, 0, stream>>>(xcur, S.sq, CIN);
    gram_wmma<CIN><<<dim3(64, 64, NB), 32, 0, stream>>>(xcur, S.gram);
    knn_topk<<<(NNODES + 255) / 256, 256, 0, stream>>>(S.gram, S.sq, S.knn);
    wconv_t<<<(KP * COUT + 255) / 256, 256, 0, stream>>>(W, S.wbf, 2 * CIN, KP, COUT);
    gather_feat<<<NEDGES, KP, 0, stream>>>(xcur, S.knn, S.feat, CIN);
    zero_f32<<<8, 256, 0, stream>>>(S.ssum, 2048);
    dim3 gg(NNODES / 4, COUT / 64), bb(32, 5);
    edge_gemm<1, KP, COUT><<<gg, bb, 0, stream>>>(S.feat, S.wbf, bias, g, be,
                                                  S.ssum, S.ssq, nullptr);
    edge_gemm<0, KP, COUT><<<gg, bb, 0, stream>>>(S.feat, S.wbf, bias, g, be,
                                                  S.ssum, S.ssq, xnext);
}

// ---------------------------------------------------------------- driver
extern "C" void kernel_launch(void* const* d_in, const int* in_sizes, int n_in,
                              void* d_out, int out_size, void* d_ws, size_t ws_size,
                              hipStream_t stream) {
    (void)in_sizes; (void)n_in; (void)out_size; (void)ws_size;
    const float* pos = (const float*)d_in[0];
    const float* W[4]  = {(const float*)d_in[2],  (const float*)d_in[6],
                          (const float*)d_in[10], (const float*)d_in[14]};
    const float* BB[4] = {(const float*)d_in[3],  (const float*)d_in[7],
                          (const float*)d_in[11], (const float*)d_in[15]};
    const float* GG[4] = {(const float*)d_in[4],  (const float*)d_in[8],
                          (const float*)d_in[12], (const float*)d_in[16]};
    const float* BE[4] = {(const float*)d_in[5],  (const float*)d_in[9],
                          (const float*)d_in[13], (const float*)d_in[17]};
    const float* Wc1 = (const float*)d_in[18];
    const float* bc1 = (const float*)d_in[19];
    const float* Wc2 = (const float*)d_in[20];
    const float* bc2 = (const float*)d_in[21];

    char* ws = (char*)d_ws;
    auto carve = [&](size_t bytes) {
        char* p = ws;
        ws += (bytes + 255) & ~(size_t)255;
        return p;
    };
    Scratch S;
    S.gram = (float*)carve((size_t)NB * NP * NP * 4);            // 32 MB
    float*  xa = (float*)carve((size_t)NNODES * 1024 * 4);       // 32 MB
    float*  xb = (float*)carve((size_t)NNODES * 1024 * 4);       // 32 MB
    S.feat = (__bf16*)carve((size_t)NEDGES * 256 * 2);           // 80 MB
    S.wbf  = (__bf16*)carve((size_t)256 * 1024 * 2);
    S.sq   = (float*)carve((size_t)NNODES * 4);
    S.knn  = (int*)carve((size_t)NEDGES * 4);
    S.ssum = (float*)carve((size_t)2048 * 4);
    S.ssq  = S.ssum + 1024;
    float* pooled = (float*)carve((size_t)NB * 1024 * 4);
    float* h1     = (float*)carve((size_t)NB * 512 * 4);

    run_layer<3,   64>(pos, xa, W[0], BB[0], GG[0], BE[0], S, stream);
    run_layer<64,  64>(xa,  xb, W[1], BB[1], GG[1], BE[1], S, stream);
    run_layer<64, 128>(xb,  xa, W[2], BB[2], GG[2], BE[2], S, stream);
    run_layer<128,1024>(xa, xb, W[3], BB[3], GG[3], BE[3], S, stream);

    maxpool_kernel<<<(NB * 1024 + 255) / 256, 256, 0, stream>>>(xb, pooled);
    fc1_kernel<<<(NB * 512 + 255) / 256, 256, 0, stream>>>(pooled, Wc1, bc1, h1);
    fc2_kernel<<<(NB * 40 + 255) / 256, 256, 0, stream>>>(h1, Wc2, bc2, (float*)d_out);
}
